// AdditiveAttention_24713241821535
// MI455X (gfx1250) — matrix operand order repaired and checked by reading, not compiled, not run on hardware
//
#include <hip/hip_runtime.h>
#include <hip/hip_bf16.h>
#include <math.h>

// ---------------------------------------------------------------------------
// AdditiveAttention for MI455X (gfx1250, wave32, WMMA)
//   B=2, N=M=1024, D=256, H=128
//   Stage 1: convert f32 -> bf16 copies (incl. transposed W and key for
//            K-contiguous WMMA B-fragment loads)
//   Stage 2: a = q@Wa + ba, b = k@Wb + bb     (v_wmma_f32_16x16x32_bf16)
//   Stage 3: scores = sum_h tanh(a+b)*v ; fused row softmax -> attn (bf16)
//            8 n-rows per block -> each b-element feeds 32 tanh+fma
//            (trans-pipe bound on v_tanh_f32; WMMA cannot help: tanh is
//            inside the H-contraction)
//   Stage 4: out = attn @ key                  (v_wmma_f32_16x16x32_bf16)
// ---------------------------------------------------------------------------

typedef __bf16 v4bf  __attribute__((ext_vector_type(4)));
typedef __bf16 v8bf  __attribute__((ext_vector_type(8)));
typedef __bf16 v16bf __attribute__((ext_vector_type(16)));
typedef float  v8f   __attribute__((ext_vector_type(8)));

#if defined(__has_builtin) && __has_builtin(__builtin_amdgcn_tanhf)
#  define TANHF(x) __builtin_amdgcn_tanhf(x)     // gfx1250 v_tanh_f32
#else
#  pragma message("no __builtin_amdgcn_tanhf; using inline-asm v_tanh_f32")
__device__ __forceinline__ float tanh_hw(float x) {
    float y;
    // v_tanh_f32 is a TRANS op: hardware does not interlock the next
    // dependent VALU op, so pad with one v_nop (ISA 15.5 multicycle rule).
    asm("v_tanh_f32 %0, %1\n\tv_nop" : "=v"(y) : "v"(x));
    return y;
}
#  define TANHF(x) tanh_hw(x)
#endif

static constexpr int Bb = 2;
static constexpr int Nn = 1024;
static constexpr int Mm = 1024;
static constexpr int Dd = 256;
static constexpr int Hh = 128;
static constexpr int TN = 8;     // n-rows per scores block

// ---------------------------------------------------------------------------
// Stage 1: f32 -> bf16 conversions + transposes.
//   qbf [B*N, D]        row-major       (WMMA A operand, proj-a)
//   kbf [B*M, D]        row-major       (WMMA A operand, proj-b)
//   kT  [B][D][M]       (key transposed, WMMA B operand of output GEMM)
//   Wat [H][D], Wbt [H][D]  (W transposed, WMMA B operand of projections)
// ---------------------------------------------------------------------------
__global__ void convert_kernel(const float* __restrict__ q,
                               const float* __restrict__ k,
                               const float* __restrict__ Wa,
                               const float* __restrict__ Wb,
                               __bf16* __restrict__ qbf,
                               __bf16* __restrict__ kbf,
                               __bf16* __restrict__ kT,
                               __bf16* __restrict__ Wat,
                               __bf16* __restrict__ Wbt)
{
    const int i = blockIdx.x * blockDim.x + threadIdx.x;   // 0 .. B*N*D-1
    qbf[i] = (__bf16)q[i];
    const float kv = k[i];
    kbf[i] = (__bf16)kv;

    // scatter key into [B][D][M] transposed layout
    const int b = i >> 18;            // / (M*D)
    const int r = i & 262143;         // % (M*D)
    const int m = r >> 8;             // / D
    const int d = r & 255;            // % D
    kT[(size_t)b * (Dd * Mm) + (size_t)d * Mm + m] = (__bf16)kv;

    // W transposes: H*D = 32768 elements, index i = h*D + d
    if (i < Hh * Dd) {
        const int hh = i >> 8;
        const int dd = i & 255;
        Wat[i] = (__bf16)Wa[dd * Hh + hh];
        Wbt[i] = (__bf16)Wb[dd * Hh + hh];
    }
}

// ---------------------------------------------------------------------------
// Generic bf16 WMMA GEMM:  C[tM*16.., tN*16..] = A(row-major, lda) x B + bias
// B supplied TRANSPOSED as Bt[N][K] so fragment loads are K-contiguous.
// One wave computes one 16x16 f32 tile; K stepped by 32 (16x16x32 bf16 WMMA).
//
// Fragment addressing (wave32; same contiguous k order in A and B):
//   A (16x32 bf16): lanes 0-15 hold row=lane, K=0..15;
//                   lanes 16-31 hold row=lane-16, K=16..31.
//   B (32x16 bf16): col=lane&15; lanes 0-15 K=0..15, lanes 16-31 K=16..31.
//   C/D (16x16 f32): col=lane&15; VGPR j -> row = j + (lane<16 ? 0 : 8).
// Grid is sized exactly; no divergence -> EXEC all-1s at every WMMA.
// ---------------------------------------------------------------------------
__global__ void wmma_gemm_bf16(const __bf16* __restrict__ A,  int lda, long sA,
                               const __bf16* __restrict__ Bt, int ldb, long sB,
                               float* __restrict__ C,         int ldc, long sC,
                               const float* __restrict__ bias, int K, int ntN)
{
    const int lane = threadIdx.x & 31;
    const int wave = blockIdx.x * (blockDim.x >> 5) + (threadIdx.x >> 5);
    const int tM   = wave / ntN;
    const int tN   = wave % ntN;
    const int g    = lane >> 4;       // lane group (0: lanes 0-15, 1: 16-31)
    const int l    = lane & 15;

    A  += (long)blockIdx.y * sA;
    Bt += (long)blockIdx.y * sB;
    C  += (long)blockIdx.y * sC;

    const __bf16* ap = A  + (size_t)(tM * 16 + l) * lda + g * 16;
    const __bf16* bp = Bt + (size_t)(tN * 16 + l) * ldb + g * 16;

    v8f acc = {};
    for (int k0 = 0; k0 < K; k0 += 32) {
        // A fragment: two 16B K-contiguous chunks (b128 loads)
        v8bf a0 = *(const v8bf*)(ap + k0);        // K = k0 + g*16 + 0..7
        v8bf a1 = *(const v8bf*)(ap + k0 + 8);    // K = k0+8+g*16 + 0..7
        v16bf av = __builtin_shufflevector(a0, a1,
                       0, 1, 2, 3, 4, 5, 6, 7, 8, 9, 10, 11, 12, 13, 14, 15);
        // B fragment: 16 K-contiguous bf16 (32B, two b128 loads)
        v16bf bv = *(const v16bf*)(bp + k0);

        acc = __builtin_amdgcn_wmma_f32_16x16x32_bf16(
                  false, av, false, bv, (short)0, acc, false, false);
    }

    const int   col  = tN * 16 + l;
    const float bval = bias ? bias[col] : 0.0f;
    float* cp = C + (size_t)(tM * 16 + g * 8) * ldc + col;
#pragma unroll
    for (int j = 0; j < 8; ++j)
        cp[(size_t)j * ldc] = acc[j] + bval;
}

// ---------------------------------------------------------------------------
// Stage 3: fused scores + softmax, TN=8 n-rows per block.
// Block (256 threads) owns rows n0..n0+7 of one batch. Thread t owns columns
// m = 4t..4t+3. Every loaded b[m, h..h+3] float4 feeds 8 rows x 4 h = 32
// tanh+fma (a-rows live in LDS -> registers), so L2 traffic for bproj drops
// 8x to ~128 MB total while the 268M v_tanh_f32 (the true floor) stay put.
// Then 8 sequential block-wide softmax reductions; attn written as bf16
// (A operand of the stage-4 WMMA GEMM).
// ---------------------------------------------------------------------------
__global__ void scores_softmax_kernel(const float* __restrict__ aproj,
                                      const float* __restrict__ bproj,
                                      const float* __restrict__ vw,
                                      __bf16* __restrict__ attnbf)
{
    const int n0 = blockIdx.x * TN;
    const int b  = blockIdx.y;
    const int t  = threadIdx.x;          // 0..255

    __shared__ float a_sh[TN * Hh];      // 8 rows of a (contiguous in aproj)
    __shared__ float v_sh[Hh];
    __shared__ float red[256];

    const float* ablk = aproj + ((size_t)b * Nn + n0) * Hh;
#pragma unroll
    for (int i = t; i < TN * Hh; i += 256) a_sh[i] = ablk[i];
    if (t < Hh) v_sh[t] = vw[t];
    __syncthreads();

    const float* bbase = bproj + (size_t)b * Mm * Hh;
    const float4* brow[4];
#pragma unroll
    for (int j = 0; j < 4; ++j)
        brow[j] = (const float4*)(bbase + (size_t)(t * 4 + j) * Hh);

    float acc[TN][4];
#pragma unroll
    for (int nn = 0; nn < TN; ++nn)
#pragma unroll
        for (int j = 0; j < 4; ++j) acc[nn][j] = 0.0f;

    for (int h4 = 0; h4 < Hh / 4; ++h4) {
        const float4 vv = *(const float4*)&v_sh[h4 * 4];
        float4 a4[TN];
#pragma unroll
        for (int nn = 0; nn < TN; ++nn)
            a4[nn] = *(const float4*)&a_sh[nn * Hh + h4 * 4];
#pragma unroll
        for (int j = 0; j < 4; ++j) {
            const float4 bv = brow[j][h4];
#pragma unroll
            for (int nn = 0; nn < TN; ++nn) {
                acc[nn][j] += TANHF(a4[nn].x + bv.x) * vv.x;
                acc[nn][j] += TANHF(a4[nn].y + bv.y) * vv.y;
                acc[nn][j] += TANHF(a4[nn].z + bv.z) * vv.z;
                acc[nn][j] += TANHF(a4[nn].w + bv.w) * vv.w;
            }
        }
    }

    // per-row softmax over M=1024 (4 values per thread), rows sequentially
    for (int nn = 0; nn < TN; ++nn) {
        float lmax = fmaxf(fmaxf(acc[nn][0], acc[nn][1]),
                           fmaxf(acc[nn][2], acc[nn][3]));
        red[t] = lmax;
        __syncthreads();
        for (int off = 128; off > 0; off >>= 1) {
            if (t < off) red[t] = fmaxf(red[t], red[t + off]);
            __syncthreads();
        }
        const float rmax = red[0];
        __syncthreads();

        float e[4];
        float lsum = 0.0f;
#pragma unroll
        for (int j = 0; j < 4; ++j) {
            e[j] = __expf(acc[nn][j] - rmax);
            lsum += e[j];
        }
        red[t] = lsum;
        __syncthreads();
        for (int off = 128; off > 0; off >>= 1) {
            if (t < off) red[t] += red[t + off];
            __syncthreads();
        }
        const float inv = 1.0f / red[0];
        __syncthreads();                 // red reused next iteration

        v4bf o;
#pragma unroll
        for (int j = 0; j < 4; ++j) o[j] = (__bf16)(e[j] * inv);
        *(v4bf*)(attnbf + ((size_t)b * Nn + n0 + nn) * Mm + t * 4) = o;
    }
}

// ---------------------------------------------------------------------------
extern "C" void kernel_launch(void* const* d_in, const int* in_sizes, int n_in,
                              void* d_out, int out_size, void* d_ws, size_t ws_size,
                              hipStream_t stream)
{
    (void)in_sizes; (void)n_in; (void)out_size; (void)ws_size;

    const float* q   = (const float*)d_in[0];   // [B,N,D]
    const float* k   = (const float*)d_in[1];   // [B,M,D]
    const float* Waw = (const float*)d_in[2];   // [D,H]
    const float* Wab = (const float*)d_in[3];   // [H]
    const float* Wbw = (const float*)d_in[4];   // [D,H]
    const float* Wbb = (const float*)d_in[5];   // [H]
    const float* vw  = (const float*)d_in[6];   // [H]
    float* out = (float*)d_out;                 // [B,N,D]

    // workspace carve-up (~9.6 MB total)
    char*  ws  = (char*)d_ws;
    size_t off = 0;
    auto alloc = [&](size_t bytes) -> char* {
        char* p = ws + off;
        off = (off + bytes + 255) & ~(size_t)255;
        return p;
    };
    __bf16* qbf    = (__bf16*)alloc((size_t)Bb * Nn * Dd * sizeof(__bf16)); // 1 MB
    __bf16* kbf    = (__bf16*)alloc((size_t)Bb * Mm * Dd * sizeof(__bf16)); // 1 MB
    __bf16* kT     = (__bf16*)alloc((size_t)Bb * Dd * Mm * sizeof(__bf16)); // 1 MB
    __bf16* Wat    = (__bf16*)alloc((size_t)Hh * Dd * sizeof(__bf16));      // 64 KB
    __bf16* Wbt    = (__bf16*)alloc((size_t)Hh * Dd * sizeof(__bf16));      // 64 KB
    float*  aproj  = (float*)alloc((size_t)Bb * Nn * Hh * sizeof(float));   // 1 MB
    float*  bproj  = (float*)alloc((size_t)Bb * Mm * Hh * sizeof(float));   // 1 MB
    __bf16* attnbf = (__bf16*)alloc((size_t)Bb * Nn * Mm * sizeof(__bf16)); // 4 MB

    // Stage 1: conversions / transposes (B*N*D = 524288 threads)
    convert_kernel<<<(Bb * Nn * Dd) / 256, 256, 0, stream>>>(
        q, k, Waw, Wbw, qbf, kbf, kT, Wat, Wbt);

    // Stage 2: projections. Tiles = (B*N/16)*(H/16) = 128*8 = 1024 waves
    //          -> 128 blocks of 256 threads (8 waves/block).
    wmma_gemm_bf16<<<dim3(128, 1), 256, 0, stream>>>(
        qbf, Dd, 0L, Wat, Dd, 0L, aproj, Hh, 0L, Wab, Dd, Hh / 16);
    wmma_gemm_bf16<<<dim3(128, 1), 256, 0, stream>>>(
        kbf, Dd, 0L, Wbt, Dd, 0L, bproj, Hh, 0L, Wbb, Dd, Hh / 16);

    // Stage 3: fused scores + softmax (one block per 8 n-rows per batch)
    scores_softmax_kernel<<<dim3(Nn / TN, Bb), 256, 0, stream>>>(
        aproj, bproj, vw, attnbf);

    // Stage 4: out = attn @ key. Per batch: (N/16)*(D/16) = 64*16 = 1024 waves
    //          -> 128 blocks of 256 threads, grid.y = batch.
    wmma_gemm_bf16<<<dim3(128, Bb), 256, 0, stream>>>(
        attnbf, Mm, (long)Nn * Mm,
        kT,     Mm, (long)Dd * Mm,
        out,    Dd, (long)Nn * Dd,
        nullptr, Mm, Dd / 16);
}
